// PtrNet_44332652429982
// MI455X (gfx1250) — compile-verified, matrix operand order checked by reading, not run on hardware
//
#include <hip/hip_runtime.h>
#include <hip/hip_bf16.h>
#include <math.h>

// ---------------- CDNA5 WMMA types ----------------
typedef __bf16 bf16_t;
typedef __attribute__((ext_vector_type(16))) __bf16 v16bf;
typedef __attribute__((ext_vector_type(8)))  float  v8f;

#define PINF 1e30f
constexpr int Bv = 128;
constexpr int Lv = 2048;
constexpr int Hv = 75;      // hidden
constexpr int Dv = 150;     // match dim
constexpr int UD = 225;     // H + D (W0 row stride)
constexpr int BL = Bv * Lv;          // 262144 rows
constexpr int MTILES = BL / 16;      // 16384
constexpr int WAVES_PER_BLOCK = 4;   // 128 threads

// ---------------- branch-free transcendentals (v_exp_f32 / v_rcp_f32) ------
__device__ __forceinline__ float fast_tanh(float x) {
    // tanh(x) = 1 - 2/(exp2(2x*log2e) + 1); saturates correctly for |x| large
    const float e = __builtin_amdgcn_exp2f(x * 2.885390081777927f);
    return 1.0f - 2.0f * __builtin_amdgcn_rcpf(e + 1.0f);
}
__device__ __forceinline__ float fast_exp(float x) {
    return __builtin_amdgcn_exp2f(x * 1.4426950408889634f);
}
__device__ __forceinline__ float fast_sigmoid(float x) {
    return __builtin_amdgcn_rcpf(1.0f + __builtin_amdgcn_exp2f(-x * 1.4426950408889634f));
}

// ---------------------------------------------------------------------------
// Pack W0's match columns (W0[h, 75+k]) into the per-lane bf16 B-operand
// register image for v_wmma_f32_16x16x32_bf16.  5 N-tiles x 5 K-tiles.
// B layout (16-bit, 32x16 KxN): lane holds column N=lane%16; lanes 0-15 carry
// K {kb..kb+7, kb+16..kb+23}, lanes 16-31 carry K {kb+8.., kb+24..} (kb=kt*32).
// ---------------------------------------------------------------------------
__global__ __launch_bounds__(32) void prep_b_kernel(const float* __restrict__ W0,
                                                    bf16_t* __restrict__ bpack) {
    const int lane = threadIdx.x & 31;
    const int ln   = lane & 15;
    const int koff = (lane >> 4) * 8;
    for (int nt = 0; nt < 5; ++nt) {
        const int  h  = nt * 16 + ln;
        const bool hv = (h < Hv);
        const float* brow = W0 + (hv ? h : 0) * UD + Hv;  // clamped, no OOB
        for (int kt = 0; kt < 5; ++kt) {
            const int kb = kt * 32 + koff;
            v16bf bm;
            #pragma unroll
            for (int i = 0; i < 8; ++i) {
                const int k0 = kb + i;
                const int k1 = kb + 16 + i;
                bm[i]     = (hv && k0 < Dv) ? (bf16_t)brow[k0] : (bf16_t)0.0f;
                bm[8 + i] = (hv && k1 < Dv) ? (bf16_t)brow[k1] : (bf16_t)0.0f;
            }
            *(v16bf*)(bpack + ((nt * 5 + kt) * 32 + lane) * 16) = bm;
        }
    }
}

// ---------------------------------------------------------------------------
// sp[b,h] = b0[h] + sum_j state[b,j] * W0[h,j]   (state part of the pointer)
// ---------------------------------------------------------------------------
__global__ __launch_bounds__(96) void sp_kernel(const float* __restrict__ state,
                                                const float* __restrict__ W0,
                                                const float* __restrict__ b0,
                                                float* __restrict__ sp) {
    const int b = blockIdx.x;
    const int h = threadIdx.x;
    __shared__ float st[Hv];
    if (h < Hv) st[h] = state[b * Hv + h];
    __syncthreads();
    if (h < Hv) {
        float acc = b0[h];
        const float* wr = W0 + h * UD;       // columns 0..74 = state part
        #pragma unroll 5
        for (int j = 0; j < Hv; ++j) acc += st[j] * wr[j];
        sp[b * 80 + h] = acc;
    } else if (h < 80) {
        sp[b * 80 + h] = 0.0f;
    }
}

// ---------------------------------------------------------------------------
// Fused pointer pass: per 16-row M-tile (one wave):
//   acc[m,h] = match[m,:] @ W0m[h,:]   (5x5 bf16 WMMAs, f32 accum)
//   logit[m] = b1 + sum_h tanh(acc + sp[b,h]) * W1[h]  - INF*(1-mask)
// ---------------------------------------------------------------------------
__global__ __launch_bounds__(WAVES_PER_BLOCK * 32)
void ptr_logits_kernel(const float* __restrict__ match,
                       const bf16_t* __restrict__ bpack,
                       const float* __restrict__ W1,
                       const float* __restrict__ b1,
                       const unsigned char* __restrict__ mask,
                       const float* __restrict__ sp,
                       float* __restrict__ out) {
    const int lane  = threadIdx.x & 31;
    const int wave  = threadIdx.x >> 5;
    const int mtile = blockIdx.x * WAVES_PER_BLOCK + wave;
    const int m0    = mtile * 16;
    const int b     = m0 >> 11;           // / L ; 16-row tile never crosses b
    const int half  = lane >> 4;
    const int ln    = lane & 15;
    const int koff  = half * 8;

    // --- load + convert the A tile (16 rows x 150 f32 -> 5 bf16 K-tiles) ---
    // Rows are 8-byte aligned (stride 600B) and every K-group starts at a
    // multiple of 8 floats -> float2 loads.  Only the last group of K-tile 4
    // (K=144..159) crosses K=150 and needs guarded scalar loads.
    const float* arow = match + (size_t)(m0 + ln) * Dv;
    v16bf A[5];
    #pragma unroll
    for (int kt = 0; kt < 5; ++kt) {
        const int kb = kt * 32 + koff;
        {   // group 0: K = kb .. kb+7  (kb <= 136, so always < 150)
            const float2* p = (const float2*)(arow + kb);
            const float2 q0 = p[0], q1 = p[1], q2 = p[2], q3 = p[3];
            A[kt][0] = (bf16_t)q0.x; A[kt][1] = (bf16_t)q0.y;
            A[kt][2] = (bf16_t)q1.x; A[kt][3] = (bf16_t)q1.y;
            A[kt][4] = (bf16_t)q2.x; A[kt][5] = (bf16_t)q2.y;
            A[kt][6] = (bf16_t)q3.x; A[kt][7] = (bf16_t)q3.y;
        }
        if (kt < 4) {   // group 1: K = kb+16 .. kb+23, fully valid
            const float2* p = (const float2*)(arow + kb + 16);
            const float2 q0 = p[0], q1 = p[1], q2 = p[2], q3 = p[3];
            A[kt][ 8] = (bf16_t)q0.x; A[kt][ 9] = (bf16_t)q0.y;
            A[kt][10] = (bf16_t)q1.x; A[kt][11] = (bf16_t)q1.y;
            A[kt][12] = (bf16_t)q2.x; A[kt][13] = (bf16_t)q2.y;
            A[kt][14] = (bf16_t)q3.x; A[kt][15] = (bf16_t)q3.y;
        } else {        // tail group straddles K=150: guarded scalar loads
            #pragma unroll
            for (int i = 0; i < 8; ++i) {
                const int k = kb + 16 + i;
                A[kt][8 + i] = (k < Dv) ? (bf16_t)arow[k] : (bf16_t)0.0f;
            }
        }
    }

    float plog[8] = {0.f, 0.f, 0.f, 0.f, 0.f, 0.f, 0.f, 0.f};
    const float b1v = b1[0];

    #pragma unroll
    for (int nt = 0; nt < 5; ++nt) {
        const int  h  = nt * 16 + ln;
        const bool hv = (h < Hv);
        v8f acc = {0.f, 0.f, 0.f, 0.f, 0.f, 0.f, 0.f, 0.f};
        #pragma unroll
        for (int kt = 0; kt < 5; ++kt) {
            v16bf Bm = *(const v16bf*)(bpack + ((nt * 5 + kt) * 32 + lane) * 16);
            acc = __builtin_amdgcn_wmma_f32_16x16x32_bf16(
                      false, A[kt], false, Bm, (short)0, acc, false, false);
        }
        const float w1h = hv ? W1[h]          : 0.0f;
        const float sph = hv ? sp[b * 80 + h] : 0.0f;
        #pragma unroll
        for (int r = 0; r < 8; ++r)
            plog[r] += w1h * fast_tanh(acc[r] + sph);
    }

    // reduce over h: sum across the 16 lanes of each half-wave (xor < 16
    // never crosses the lane-16 boundary, matching the f32 C/D layout)
    #pragma unroll
    for (int r = 0; r < 8; ++r) {
        float v = plog[r];
        v += __shfl_xor(v, 1);
        v += __shfl_xor(v, 2);
        v += __shfl_xor(v, 4);
        v += __shfl_xor(v, 8);
        plog[r] = v;
    }

    if (ln == 0) {
        #pragma unroll
        for (int r = 0; r < 8; ++r) {
            const int row = m0 + half * 8 + r;
            const float mk = mask[row] ? 1.0f : 0.0f;
            out[row] = plog[r] + b1v - PINF * (1.0f - mk);
        }
    }
}

// ---------------------------------------------------------------------------
// Softmax over L per batch + weighted sum over match: res[b,d]
// ---------------------------------------------------------------------------
__global__ __launch_bounds__(256) void attn_kernel(const float* __restrict__ logits,
                                                   const float* __restrict__ match,
                                                   float* __restrict__ res) {
    const int b   = blockIdx.x;
    const int tid = threadIdx.x;
    __shared__ float w[Lv];
    __shared__ float red[256];
    const float* lg = logits + (size_t)b * Lv;

    float mx = -PINF;
    for (int l = tid; l < Lv; l += 256) mx = fmaxf(mx, lg[l]);
    red[tid] = mx;
    __syncthreads();
    for (int s = 128; s > 0; s >>= 1) {
        if (tid < s) red[tid] = fmaxf(red[tid], red[tid + s]);
        __syncthreads();
    }
    mx = red[0];
    __syncthreads();

    float sum = 0.0f;
    for (int l = tid; l < Lv; l += 256) {
        const float e = fast_exp(lg[l] - mx);
        w[l] = e;
        sum += e;
    }
    red[tid] = sum;
    __syncthreads();
    for (int s = 128; s > 0; s >>= 1) {
        if (tid < s) red[tid] += red[tid + s];
        __syncthreads();
    }
    const float inv = __builtin_amdgcn_rcpf(red[0]);
    __syncthreads();

    if (tid < Dv) {
        float acc = 0.0f;
        const float* mb = match + (size_t)b * Lv * Dv + tid;
        for (int l = 0; l < Lv; ++l) acc += w[l] * mb[(size_t)l * Dv];
        res[b * Dv + tid] = acc * inv;
    }
}

// ---------------------------------------------------------------------------
// One GRU step with h0 = 0:  h = (1-z)*n,  gh = b_hh
// ---------------------------------------------------------------------------
__global__ __launch_bounds__(256) void gru_kernel(const float* __restrict__ res,
                                                  const float* __restrict__ w_ih,
                                                  const float* __restrict__ b_ih,
                                                  const float* __restrict__ b_hh,
                                                  float* __restrict__ hstate) {
    const int b   = blockIdx.x;
    const int tid = threadIdx.x;
    __shared__ float x[Dv];
    __shared__ float gi[3 * Hv];
    if (tid < Dv) x[tid] = res[b * Dv + tid];
    __syncthreads();
    if (tid < 3 * Hv) {
        float acc = b_ih[tid];
        const float* wr = w_ih + tid * Dv;
        #pragma unroll 5
        for (int d = 0; d < Dv; ++d) acc += x[d] * wr[d];
        gi[tid] = acc;
    }
    __syncthreads();
    if (tid < Hv) {
        const float r = fast_sigmoid(gi[tid] + b_hh[tid]);
        const float z = fast_sigmoid(gi[Hv + tid] + b_hh[Hv + tid]);
        const float n = fast_tanh(gi[2 * Hv + tid] + r * b_hh[2 * Hv + tid]);
        hstate[b * Hv + tid] = (1.0f - z) * n;
    }
}

// ---------------------------------------------------------------------------
extern "C" void kernel_launch(void* const* d_in, const int* in_sizes, int n_in,
                              void* d_out, int out_size, void* d_ws, size_t ws_size,
                              hipStream_t stream) {
    const float*         init  = (const float*)d_in[0];          // [B,H]
    const float*         match = (const float*)d_in[1];          // [B,L,D]
    const unsigned char* cmask = (const unsigned char*)d_in[2];  // [B,L] bool
    const float*         W0    = (const float*)d_in[3];          // [H, H+D]
    const float*         b0    = (const float*)d_in[4];          // [H]
    const float*         W1    = (const float*)d_in[5];          // [1,H]
    const float*         b1    = (const float*)d_in[6];          // [1]
    const float*         w_ih  = (const float*)d_in[7];          // [3H, D]
    /* w_hh (d_in[8]) unused: h0 == 0  -> gh = b_hh */
    const float*         b_ih  = (const float*)d_in[9];          // [3H]
    const float*         b_hh  = (const float*)d_in[10];         // [3H]
    float* out = (float*)d_out;                                  // [2, B, L]

    // workspace carve-up (all small): packed B operands, sp, res, hstate
    bf16_t* bpack = (bf16_t*)d_ws;                        // 5*5*32*16 bf16 = 25.6 KB
    float*  sp    = (float*)(bpack + 5 * 5 * 32 * 16);    // [B,80]
    float*  res   = sp + Bv * 80;                         // [B,D]
    float*  hst   = res + Bv * Dv;                        // [B,H]

    prep_b_kernel<<<1, 32, 0, stream>>>(W0, bpack);

    // ---- pointer pass 1: state = init ----
    sp_kernel<<<Bv, 96, 0, stream>>>(init, W0, b0, sp);
    ptr_logits_kernel<<<MTILES / WAVES_PER_BLOCK, WAVES_PER_BLOCK * 32, 0, stream>>>(
        match, bpack, W1, b1, cmask, sp, out);

    // ---- attention-weighted input + GRU step ----
    attn_kernel<<<Bv, 256, 0, stream>>>(out, match, res);
    gru_kernel<<<Bv, 256, 0, stream>>>(res, w_ih, b_ih, b_hh, hst);

    // ---- pointer pass 2: state = GRU output ----
    sp_kernel<<<Bv, 96, 0, stream>>>(hst, W0, b0, sp);
    ptr_logits_kernel<<<MTILES / WAVES_PER_BLOCK, WAVES_PER_BLOCK * 32, 0, stream>>>(
        match, bpack, W1, b1, cmask, sp, out + BL);
}
